// ContinuousField_90417651515903
// MI455X (gfx1250) — matrix operand order repaired and checked by reading, not compiled, not run on hardware
//
#include <hip/hip_runtime.h>
#include <math.h>

typedef __bf16 v16bf __attribute__((ext_vector_type(16)));
typedef float  v8f   __attribute__((ext_vector_type(8)));

static constexpr int Bn   = 8;
static constexpr int Sn   = 16384;
static constexpr int IND  = 256;   // IN_DIM
static constexpr int FLD  = 256;   // FIELD
static constexpr int OD   = 321;   // OUT_DIM
static constexpr int KENT = 2048;
static constexpr int POSD = 64;

static constexpr int RS_F = 264;   // 256-col LDS staging stride (bf16 elems)
static constexpr int RS_H = 520;   // 512-col LDS staging stride

union Frag16 { v16bf v; unsigned int u[8]; };
union Acc8   { v8f v; float f[8]; };

__device__ __forceinline__ unsigned int pk2bf(float a, float b) {
  union { __bf16 e[2]; unsigned int u; } c;
  c.e[0] = (__bf16)a; c.e[1] = (__bf16)b;
  return c.u;
}
__device__ __forceinline__ float gelu_exact(float v) {
  return 0.5f * v * (1.0f + erff(v * 0.70710678118654752f));
}
__device__ __forceinline__ v8f wmma_bf16(v16bf a, v16bf b, v8f c) {
  return __builtin_amdgcn_wmma_f32_16x16x32_bf16(false, a, false, b, (short)0, c, false, false);
}

// B-matrix fragment: 32 lanes x 32B contiguous, pre-packed by prep kernel.
__device__ __forceinline__ v16bf load_bfrag(const unsigned short* wf, int frag, int lane) {
  const uint4* p = (const uint4*)(wf + (size_t)frag * 512 + lane * 16);
  uint4 q0 = p[0], q1 = p[1];
  Frag16 f;
  f.u[0] = q0.x; f.u[1] = q0.y; f.u[2] = q0.z; f.u[3] = q0.w;
  f.u[4] = q1.x; f.u[5] = q1.y; f.u[6] = q1.z; f.u[7] = q1.w;
  return f.v;
}

// A-matrix 16x32 fragment from global fp32 row-major (rowPtr -> this lane's row).
__device__ __forceinline__ v16bf load_afrag_global(const float* rowPtr, int kb, int lane) {
  int hi = (lane & 16) ? 8 : 0;
  const float4* p0 = (const float4*)(rowPtr + kb + hi);
  const float4* p1 = (const float4*)(rowPtr + kb + 16 + hi);
  float4 a0 = p0[0], a1 = p0[1], b0 = p1[0], b1 = p1[1];
  Frag16 f;
  f.u[0] = pk2bf(a0.x, a0.y); f.u[1] = pk2bf(a0.z, a0.w);
  f.u[2] = pk2bf(a1.x, a1.y); f.u[3] = pk2bf(a1.z, a1.w);
  f.u[4] = pk2bf(b0.x, b0.y); f.u[5] = pk2bf(b0.z, b0.w);
  f.u[6] = pk2bf(b1.x, b1.y); f.u[7] = pk2bf(b1.z, b1.w);
  return f.v;
}

// A-matrix 16x32 fragment from LDS bf16 staging buffer (row stride RS elems).
__device__ __forceinline__ v16bf load_afrag_lds(const unsigned short* buf, int RS, int kb, int lane) {
  int m  = lane & 15;
  int hi = (lane & 16) ? 8 : 0;
  const uint4* p0 = (const uint4*)(buf + m * RS + kb + hi);
  const uint4* p1 = (const uint4*)(buf + m * RS + kb + 16 + hi);
  uint4 q0 = *p0, q1 = *p1;
  Frag16 f;
  f.u[0] = q0.x; f.u[1] = q0.y; f.u[2] = q0.z; f.u[3] = q0.w;
  f.u[4] = q1.x; f.u[5] = q1.y; f.u[6] = q1.z; f.u[7] = q1.w;
  return f.v;
}

// C fragment (16x16 f32) -> LDS bf16 staging, column block colBase.
__device__ __forceinline__ void store_cfrag_lds(unsigned short* buf, int RS, int colBase, int lane, const Acc8& c) {
  __bf16* bb = (__bf16*)buf;
  int n  = colBase + (lane & 15);
  int r0 = (lane & 16) ? 8 : 0;
#pragma unroll
  for (int v = 0; v < 8; ++v) bb[(r0 + v) * RS + n] = (__bf16)c.f[v];
}

// ---------------- weight fragment packing ----------------
__global__ __launch_bounds__(256) void prep_weights(
    const float* W_in, const float* W1, const float* W2, const float* W_sp,
    unsigned short* winF, unsigned short* w1F, unsigned short* w2F, unsigned short* wspF) {
  int slot = blockIdx.x * blockDim.x + threadIdx.x;
  const float* W; unsigned short* O; int KT, N; int rel;
  if      (slot <  4096) { W = W_in; O = winF; KT = 8;  N = 256; rel = slot; }
  else if (slot < 12288) { W = W1;   O = w1F;  KT = 8;  N = 512; rel = slot - 4096; }
  else if (slot < 23040) { W = W2;   O = w2F;  KT = 16; N = 321; rel = slot - 12288; }
  else if (slot < 27136) { W = W_sp; O = wspF; KT = 8;  N = 256; rel = slot - 23040; }
  else return;
  int lane = rel & 31;
  int frag = rel >> 5;
  int kt = frag % KT;
  int nt = frag / KT;
  int n  = nt * 16 + (lane & 15);
  int kb = kt * 32 + ((lane & 16) ? 8 : 0);
  __bf16* dst = (__bf16*)(O + (size_t)frag * 512 + lane * 16);
#pragma unroll
  for (int j = 0; j < 16; ++j) {
    int k = kb + j + ((j >= 8) ? 8 : 0);
    float v = (n < N) ? W[(size_t)k * N + n] : 0.0f;
    dst[j] = (__bf16)v;
  }
}

// ---------------- Phase A: fused MLP -> per-token feature norm ----------------
// One wave (32 threads) per block; each wave handles TWO 16-token M-tiles so
// every weight B-fragment feeds two WMMAs (halves L2 weight traffic).
__global__ __launch_bounds__(32) void phaseA(
    const float* __restrict__ x,
    const unsigned short* __restrict__ winF, const float* __restrict__ bIn,
    const unsigned short* __restrict__ w1F,  const float* __restrict__ b1,
    const unsigned short* __restrict__ w2F,  const float* __restrict__ b2,
    float* __restrict__ norms) {
  __shared__ unsigned short smem[2 * 16 * RS_H];  // two h-staging buffers (33 KB)
  int lane = threadIdx.x & 31;
  unsigned short* h0 = smem;
  unsigned short* h1 = smem + 16 * RS_H;
  int tp    = blockIdx.x;          // tile pair
  int b     = tp >> 9;             // 512 pairs per batch
  int sbase = (tp & 511) << 5;     // 32 tokens
  const float* xr0 = x + ((size_t)(b * Sn + sbase + (lane & 15))) * IND;
  const float* xr1 = xr0 + (size_t)16 * IND;

  // GEMM1: field = x @ W_in + b_in
  v16bf xa0[8], xa1[8];
#pragma unroll
  for (int kt = 0; kt < 8; ++kt) { xa0[kt] = load_afrag_global(xr0, kt * 32, lane);
                                   xa1[kt] = load_afrag_global(xr1, kt * 32, lane); }
  for (int nt = 0; nt < 16; ++nt) {
    float bias = bIn[nt * 16 + (lane & 15)];
    Acc8 c0, c1;
#pragma unroll
    for (int i = 0; i < 8; ++i) { c0.f[i] = bias; c1.f[i] = bias; }
#pragma unroll
    for (int kt = 0; kt < 8; ++kt) {
      v16bf bf = load_bfrag(winF, nt * 8 + kt, lane);
      c0.v = wmma_bf16(xa0[kt], bf, c0.v);
      c1.v = wmma_bf16(xa1[kt], bf, c1.v);
    }
    store_cfrag_lds(h0, RS_F, nt * 16, lane, c0);
    store_cfrag_lds(h1, RS_F, nt * 16, lane, c1);
  }
  __syncthreads();
  v16bf fa0[8], fa1[8];
#pragma unroll
  for (int kt = 0; kt < 8; ++kt) { fa0[kt] = load_afrag_lds(h0, RS_F, kt * 32, lane);
                                   fa1[kt] = load_afrag_lds(h1, RS_F, kt * 32, lane); }
  __syncthreads();

  // GEMM2: h = gelu(field @ W1 + b1)
  for (int nt = 0; nt < 32; ++nt) {
    if (nt + 1 < 32) __builtin_prefetch(w1F + (size_t)((nt + 1) * 8) * 512, 0, 0);
    float bias = b1[nt * 16 + (lane & 15)];
    Acc8 c0, c1;
#pragma unroll
    for (int i = 0; i < 8; ++i) { c0.f[i] = bias; c1.f[i] = bias; }
#pragma unroll
    for (int kt = 0; kt < 8; ++kt) {
      v16bf bf = load_bfrag(w1F, nt * 8 + kt, lane);
      c0.v = wmma_bf16(fa0[kt], bf, c0.v);
      c1.v = wmma_bf16(fa1[kt], bf, c1.v);
    }
#pragma unroll
    for (int i = 0; i < 8; ++i) { c0.f[i] = gelu_exact(c0.f[i]); c1.f[i] = gelu_exact(c1.f[i]); }
    store_cfrag_lds(h0, RS_H, nt * 16, lane, c0);
    store_cfrag_lds(h1, RS_H, nt * 16, lane, c1);
  }
  __syncthreads();

  // GEMM3: ef = h @ W2 + b2 ; accumulate sum of squares per row
  float ssq0[8], ssq1[8];
#pragma unroll
  for (int i = 0; i < 8; ++i) { ssq0[i] = 0.0f; ssq1[i] = 0.0f; }
  for (int nt = 0; nt < 21; ++nt) {
    if (nt + 1 < 21) __builtin_prefetch(w2F + (size_t)((nt + 1) * 16) * 512, 0, 0);
    int col = nt * 16 + (lane & 15);
    float bias = (col < OD) ? b2[col] : 0.0f;
    Acc8 c0, c1;
#pragma unroll
    for (int i = 0; i < 8; ++i) { c0.f[i] = bias; c1.f[i] = bias; }
#pragma unroll
    for (int kt = 0; kt < 16; ++kt) {
      v16bf bf = load_bfrag(w2F, nt * 16 + kt, lane);
      c0.v = wmma_bf16(load_afrag_lds(h0, RS_H, kt * 32, lane), bf, c0.v);
      c1.v = wmma_bf16(load_afrag_lds(h1, RS_H, kt * 32, lane), bf, c1.v);
    }
#pragma unroll
    for (int i = 0; i < 8; ++i) { ssq0[i] += c0.f[i] * c0.f[i]; ssq1[i] += c1.f[i] * c1.f[i]; }
  }
  // cross-lane reduction via LDS scratch (overlays h0, all h reads are done)
  __syncthreads();
  float* red = (float*)smem;   // 32 lanes x 16 floats
#pragma unroll
  for (int i = 0; i < 8; ++i) { red[lane * 16 + i] = ssq0[i]; red[lane * 16 + 8 + i] = ssq1[i]; }
  __syncthreads();
  if (lane < 16) {
    int m = lane;
    int lbase = (m < 8) ? 0 : 16;
    int v = m & 7;
    float s0 = 0.0f, s1 = 0.0f;
    for (int l = 0; l < 16; ++l) { s0 += red[(lbase + l) * 16 + v]; s1 += red[(lbase + l) * 16 + 8 + v]; }
    norms[b * Sn + sbase + m]      = sqrtf(s0);
    norms[b * Sn + sbase + 16 + m] = sqrtf(s1);
  }
}

// ---------------- Phase B: sorted top-2048 per batch (softmax is monotone) ----------------
__global__ __launch_bounds__(1024) void topk_sort(const float* __restrict__ norms, unsigned int* __restrict__ topi) {
  extern __shared__ unsigned long long keys[];
  int b = blockIdx.x;
  const float* nb = norms + (size_t)b * Sn;
  for (int i = threadIdx.x; i < Sn; i += blockDim.x) {
    unsigned int fb = __float_as_uint(nb[i]);  // norms >= 0 -> monotone bits
    keys[i] = ((unsigned long long)fb << 32) | (unsigned int)(~i);  // ties -> smaller idx first
  }
  __syncthreads();
  for (int size = 2; size <= Sn; size <<= 1) {
    for (int stride = size >> 1; stride > 0; stride >>= 1) {
      for (int t = threadIdx.x; t < Sn / 2; t += blockDim.x) {
        int lo = ((t & ~(stride - 1)) << 1) | (t & (stride - 1));
        int hi = lo + stride;
        bool desc = ((lo & size) == 0);
        unsigned long long a = keys[lo], c = keys[hi];
        bool sw = desc ? (a < c) : (a > c);
        if (sw) { keys[lo] = c; keys[hi] = a; }
      }
      __syncthreads();
    }
  }
  for (int i = threadIdx.x; i < KENT; i += blockDim.x)
    topi[b * KENT + i] = ~((unsigned int)(keys[i] & 0xFFFFFFFFull));
}

// ---------------- Phase C: recompute MLP on selected tokens + outputs ----------------
__global__ __launch_bounds__(32) void phaseC(
    const float* __restrict__ x, const unsigned int* __restrict__ topi,
    const unsigned short* __restrict__ winF, const float* __restrict__ bIn,
    const unsigned short* __restrict__ w1F,  const float* __restrict__ b1,
    const unsigned short* __restrict__ w2F,  const float* __restrict__ b2,
    const unsigned short* __restrict__ wspF, const float* __restrict__ bSp,
    float* __restrict__ out_pos, float* __restrict__ out_st, float* __restrict__ out_w) {
  __shared__ unsigned short smem[2 * 16 * RS_H + 2 * 16 * RS_F];  // 50 KB
  int lane = threadIdx.x & 31;
  unsigned short* h0 = smem;
  unsigned short* h1 = smem + 16 * RS_H;
  unsigned short* e0 = smem + 2 * 16 * RS_H;
  unsigned short* e1 = e0 + 16 * RS_F;
  int tp    = blockIdx.x;
  int b     = tp >> 6;             // 64 pairs per batch
  int tbase = (tp & 63) << 5;      // 32 selected entities
  int nloc  = lane & 15;
  unsigned int row0 = topi[b * KENT + tbase + nloc];
  unsigned int row1 = topi[b * KENT + tbase + 16 + nloc];
  const float* xr0 = x + ((size_t)b * Sn + row0) * IND;
  const float* xr1 = x + ((size_t)b * Sn + row1) * IND;

  v16bf xa0[8], xa1[8];
#pragma unroll
  for (int kt = 0; kt < 8; ++kt) { xa0[kt] = load_afrag_global(xr0, kt * 32, lane);
                                   xa1[kt] = load_afrag_global(xr1, kt * 32, lane); }
  for (int nt = 0; nt < 16; ++nt) {
    float bias = bIn[nt * 16 + nloc];
    Acc8 c0, c1;
#pragma unroll
    for (int i = 0; i < 8; ++i) { c0.f[i] = bias; c1.f[i] = bias; }
#pragma unroll
    for (int kt = 0; kt < 8; ++kt) {
      v16bf bf = load_bfrag(winF, nt * 8 + kt, lane);
      c0.v = wmma_bf16(xa0[kt], bf, c0.v);
      c1.v = wmma_bf16(xa1[kt], bf, c1.v);
    }
    store_cfrag_lds(h0, RS_F, nt * 16, lane, c0);
    store_cfrag_lds(h1, RS_F, nt * 16, lane, c1);
  }
  __syncthreads();
  v16bf fa0[8], fa1[8];
#pragma unroll
  for (int kt = 0; kt < 8; ++kt) { fa0[kt] = load_afrag_lds(h0, RS_F, kt * 32, lane);
                                   fa1[kt] = load_afrag_lds(h1, RS_F, kt * 32, lane); }
  __syncthreads();
  for (int nt = 0; nt < 32; ++nt) {
    if (nt + 1 < 32) __builtin_prefetch(w1F + (size_t)((nt + 1) * 8) * 512, 0, 0);
    float bias = b1[nt * 16 + nloc];
    Acc8 c0, c1;
#pragma unroll
    for (int i = 0; i < 8; ++i) { c0.f[i] = bias; c1.f[i] = bias; }
#pragma unroll
    for (int kt = 0; kt < 8; ++kt) {
      v16bf bf = load_bfrag(w1F, nt * 8 + kt, lane);
      c0.v = wmma_bf16(fa0[kt], bf, c0.v);
      c1.v = wmma_bf16(fa1[kt], bf, c1.v);
    }
#pragma unroll
    for (int i = 0; i < 8; ++i) { c0.f[i] = gelu_exact(c0.f[i]); c1.f[i] = gelu_exact(c1.f[i]); }
    store_cfrag_lds(h0, RS_H, nt * 16, lane, c0);
    store_cfrag_lds(h1, RS_H, nt * 16, lane, c1);
  }
  __syncthreads();

  int r0 = (lane & 16) ? 8 : 0;
  for (int nt = 0; nt < 21; ++nt) {
    if (nt + 1 < 21) __builtin_prefetch(w2F + (size_t)((nt + 1) * 16) * 512, 0, 0);
    int col = nt * 16 + nloc;
    float bias = (col < OD) ? b2[col] : 0.0f;
    Acc8 c0, c1;
#pragma unroll
    for (int i = 0; i < 8; ++i) { c0.f[i] = bias; c1.f[i] = bias; }
#pragma unroll
    for (int kt = 0; kt < 16; ++kt) {
      v16bf bf = load_bfrag(w2F, nt * 16 + kt, lane);
      c0.v = wmma_bf16(load_afrag_lds(h0, RS_H, kt * 32, lane), bf, c0.v);
      c1.v = wmma_bf16(load_afrag_lds(h1, RS_H, kt * 32, lane), bf, c1.v);
    }
    if (nt < 4) {            // positions: cols 0..63
#pragma unroll
      for (int v = 0; v < 8; ++v) {
        out_pos[((size_t)b * KENT + tbase + (r0 + v)) * POSD + col]      = c0.f[v];
        out_pos[((size_t)b * KENT + tbase + 16 + (r0 + v)) * POSD + col] = c1.f[v];
      }
    } else if (nt < 20) {    // state features: cols 64..319 -> LDS for final GEMM
      store_cfrag_lds(e0, RS_F, (nt - 4) * 16, lane, c0);
      store_cfrag_lds(e1, RS_F, (nt - 4) * 16, lane, c1);
    } else {                 // weights: col 320
      if (nloc == 0) {
#pragma unroll
        for (int v = 0; v < 8; ++v) {
          out_w[(size_t)b * KENT + tbase + (r0 + v)]      = c0.f[v];
          out_w[(size_t)b * KENT + tbase + 16 + (r0 + v)] = c1.f[v];
        }
      }
    }
  }
  __syncthreads();

  // states = sel[:,64:320] @ W_sp + b_sp
  v16bf ea0[8], ea1[8];
#pragma unroll
  for (int kt = 0; kt < 8; ++kt) { ea0[kt] = load_afrag_lds(e0, RS_F, kt * 32, lane);
                                   ea1[kt] = load_afrag_lds(e1, RS_F, kt * 32, lane); }
  for (int nt = 0; nt < 16; ++nt) {
    float bias = bSp[nt * 16 + nloc];
    Acc8 c0, c1;
#pragma unroll
    for (int i = 0; i < 8; ++i) { c0.f[i] = bias; c1.f[i] = bias; }
#pragma unroll
    for (int kt = 0; kt < 8; ++kt) {
      v16bf bf = load_bfrag(wspF, nt * 8 + kt, lane);
      c0.v = wmma_bf16(ea0[kt], bf, c0.v);
      c1.v = wmma_bf16(ea1[kt], bf, c1.v);
    }
#pragma unroll
    for (int v = 0; v < 8; ++v) {
      out_st[((size_t)b * KENT + tbase + (r0 + v)) * FLD + nt * 16 + nloc]      = c0.f[v];
      out_st[((size_t)b * KENT + tbase + 16 + (r0 + v)) * FLD + nt * 16 + nloc] = c1.f[v];
    }
  }
}

extern "C" void kernel_launch(void* const* d_in, const int* in_sizes, int n_in,
                              void* d_out, int out_size, void* d_ws, size_t ws_size,
                              hipStream_t stream) {
  const float* x    = (const float*)d_in[0];
  const float* W_in = (const float*)d_in[1];
  const float* b_in = (const float*)d_in[2];
  const float* W1   = (const float*)d_in[3];
  const float* b1   = (const float*)d_in[4];
  const float* W2   = (const float*)d_in[5];
  const float* b2   = (const float*)d_in[6];
  const float* W_sp = (const float*)d_in[7];
  const float* b_sp = (const float*)d_in[8];

  char* ws = (char*)d_ws;
  float*          norms = (float*)(ws + 0);                 // 512 KB
  unsigned int*   topi  = (unsigned int*)(ws + 524288);     // 64 KB
  unsigned short* winF  = (unsigned short*)(ws + 589824);   // 128 KB
  unsigned short* w1F   = (unsigned short*)(ws + 720896);   // 256 KB
  unsigned short* w2F   = (unsigned short*)(ws + 983040);   // 336 KB
  unsigned short* wspF  = (unsigned short*)(ws + 1327104);  // 128 KB

  float* out     = (float*)d_out;
  float* out_pos = out;                                     // [8,2048,64]
  float* out_st  = out + (size_t)Bn * KENT * POSD;          // [8,2048,256]
  float* out_w   = out_st + (size_t)Bn * KENT * FLD;        // [8,2048,1]

  prep_weights<<<106, 256, 0, stream>>>(W_in, W1, W2, W_sp, winF, w1F, w2F, wspF);
  phaseA<<<4096, 32, 0, stream>>>(x, winF, b_in, w1F, b1, w2F, b2, norms);
  topk_sort<<<8, 1024, Sn * sizeof(unsigned long long), stream>>>(norms, topi);
  phaseC<<<512, 32, 0, stream>>>(x, topi, winF, b_in, w1F, b1, w2F, b2, wspF, b_sp,
                                 out_pos, out_st, out_w);
}
